// RoPEAttention_317827580606
// MI455X (gfx1250) — matrix operand order repaired and checked by reading, not compiled, not run on hardware
//
#include <hip/hip_runtime.h>
#include <hip/hip_bf16.h>

// ---------------------------------------------------------------------------
// Problem constants (from reference): B=16, N=1029, C=1024, H=16, hd=64
// ---------------------------------------------------------------------------
#define BB   16
#define NTOK 1029
#define CC   1024
#define HH   16
#define HD   64
#define MTOT (BB * NTOK)        // 16464 rows (divisible by 16, not by 128)

typedef __attribute__((ext_vector_type(16))) __bf16 v16bf;
typedef __attribute__((ext_vector_type(8)))  __bf16 bf16x8;
typedef __attribute__((ext_vector_type(8)))  float  v8f;
typedef int v4i_gcc __attribute__((vector_size(16)));   // matches builtin param

// 16 contiguous bf16 -> fragment (B-matrix rows / packed A halves)
__device__ __forceinline__ v16bf ld16(const __bf16* p) {
    bf16x8 lo = *(const bf16x8*)p;
    bf16x8 hi = *(const bf16x8*)(p + 8);
    v16bf r;
#pragma unroll
    for (int i = 0; i < 8; ++i) { r[i] = lo[i]; r[i + 8] = hi[i]; }
    return r;
}
// two 8-element groups -> A fragment (callers: contiguous k = 16*hi + s)
__device__ __forceinline__ v16bf ld8x2(const __bf16* p0, const __bf16* p1) {
    bf16x8 lo = *(const bf16x8*)p0;
    bf16x8 hi = *(const bf16x8*)p1;
    v16bf r;
#pragma unroll
    for (int i = 0; i < 8; ++i) { r[i] = lo[i]; r[i + 8] = hi[i]; }
    return r;
}
__device__ __forceinline__ v8f wmma_bf16(v16bf a, v16bf b, v8f c) {
    // (neg_a, A, neg_b, B, c_mod, C, reuse_a, reuse_b)
    return __builtin_amdgcn_wmma_f32_16x16x32_bf16(false, a, false, b,
                                                   (short)0, c, false, false);
}

// ---------------------------------------------------------------------------
// Async global->LDS staging (CDNA5 GLOBAL_LOAD_ASYNC_TO_LDS_B128, ASYNCcnt),
// guarded so the code still compiles if the toolchain lacks the builtins.
// Probe-confirmed signature: (global v4i32*, shared v4i32*, imm offset, cpol)
// ---------------------------------------------------------------------------
#if defined(__has_builtin)
#if __has_builtin(__builtin_amdgcn_global_load_async_to_lds_b128) && \
    __has_builtin(__builtin_amdgcn_s_wait_asynccnt)
#define USE_ASYNC_LDS 1
#endif
#endif
#ifndef USE_ASYNC_LDS
#define USE_ASYNC_LDS 0
#endif

#if USE_ASYNC_LDS
__device__ __forceinline__ void stage_b128(__bf16* lds, const __bf16* g) {
    __builtin_amdgcn_global_load_async_to_lds_b128(
        (__attribute__((address_space(1))) v4i_gcc*)(void*)g,
        (__attribute__((address_space(3))) v4i_gcc*)(void*)lds, 0, 0);
}
__device__ __forceinline__ void stage_fence() {
    __builtin_amdgcn_s_wait_asynccnt(0);
}
#else
__device__ __forceinline__ void stage_b128(__bf16* lds, const __bf16* g) {
    *(bf16x8*)lds = *(const bf16x8*)g;
}
__device__ __forceinline__ void stage_fence() {}
#endif

// ---------------------------------------------------------------------------
// Cross-lane reductions WITHOUT LDS:
//  - DPP16 ROW_ROR rotations reduce over a 16-lane row (pure VALU, no waits)
//  - v_permlanex16 crosses the two 16-lane halves of a wave32
// ---------------------------------------------------------------------------
template <int CTRL>
__device__ __forceinline__ float fdpp(float x) {
    return __builtin_bit_cast(float,
        __builtin_amdgcn_update_dpp(0, __builtin_bit_cast(int, x),
                                    CTRL, 0xF, 0xF, true));
}
__device__ __forceinline__ float rowmax16(float x) {   // max over 16-lane row
    x = fmaxf(x, fdpp<0x121>(x));   // ROW_ROR:1
    x = fmaxf(x, fdpp<0x122>(x));   // ROW_ROR:2
    x = fmaxf(x, fdpp<0x124>(x));   // ROW_ROR:4
    x = fmaxf(x, fdpp<0x128>(x));   // ROW_ROR:8
    return x;
}
__device__ __forceinline__ float rowsum16(float x) {   // sum over 16-lane row
    x += fdpp<0x121>(x);
    x += fdpp<0x122>(x);
    x += fdpp<0x124>(x);
    x += fdpp<0x128>(x);
    return x;
}
__device__ __forceinline__ float xhalf(float x) {      // lane ^ 16 (wave32)
    int xi = __builtin_bit_cast(int, x);
    return __builtin_bit_cast(float,
        __builtin_amdgcn_permlanex16(xi, xi, 0x76543210, 0xFEDCBA98,
                                     false, false));
}

// ---------------------------------------------------------------------------
// Kernel 0: f32 -> bf16 conversion (grid-stride)
// ---------------------------------------------------------------------------
__global__ __launch_bounds__(256)
void cvt_bf16_kernel(const float* __restrict__ in, __bf16* __restrict__ out,
                     long long n) {
    long long i = (long long)blockIdx.x * blockDim.x + threadIdx.x;
    long long stride = (long long)gridDim.x * blockDim.x;
    for (; i < n; i += stride) out[i] = (__bf16)in[i];
}

// ---------------------------------------------------------------------------
// Kernels 1 & 4: tiled bf16 WMMA GEMM:  out[m,o] = sum_k A[m,k]*W[o,k] + bias[o]
// MODE 0: scatter bf16 into qkv buffer [3][B][H][N][64]
// MODE 1: f32 row-major out [M][1024]
// BM=128, BN=64, BK=32; 256 threads = 8 waves; wave w owns rows [16w,16w+16).
// ---------------------------------------------------------------------------
template <int MODE>
__global__ __launch_bounds__(256)
void gemm_bf16_kernel(const __bf16* __restrict__ A,   // [M][1024]
                      const __bf16* __restrict__ W,   // [O][1024]
                      const float*  __restrict__ bias,
                      void* __restrict__ out, int M) {
    constexpr int KDIM = 1024;
    constexpr int BM = 128, BN = 64, BK = 32;
    constexpr int LDA = 40;                 // padded LDS stride (16B aligned)
    __shared__ __bf16 As[BM * LDA];
    __shared__ __bf16 Ws[BN * LDA];

    const int tid  = threadIdx.x;
    const int wave = tid >> 5;
    const int lane = tid & 31;
    const int ln   = lane & 15;
    const int hi   = lane >> 4;
    const int m_blk = blockIdx.x * BM;
    const int o_blk = blockIdx.y * BN;

    v8f acc[4];
#pragma unroll
    for (int t = 0; t < 4; ++t) acc[t] = (v8f){};

    for (int k0 = 0; k0 < KDIM; k0 += BK) {
        __syncthreads();
        { // A tile: 128 rows x 32 cols; thread -> (row = tid/2, col = (tid&1)*16)
            int row = tid >> 1;
            int col = (tid & 1) * 16;
            int gm  = m_blk + row;
            __bf16* dst = &As[row * LDA + col];
            if (gm < M) {
                const __bf16* src = A + (size_t)gm * KDIM + k0 + col;
                stage_b128(dst,     src);
                stage_b128(dst + 8, src + 8);
            } else {
#pragma unroll
                for (int i = 0; i < 16; ++i) dst[i] = (__bf16)0.f;
            }
        }
        { // W tile: 64 rows x 32 cols; thread -> (row = tid/4, col = (tid&3)*8)
            int row = tid >> 2;
            int col = (tid & 3) * 8;
            const __bf16* src = W + (size_t)(o_blk + row) * KDIM + k0 + col;
            stage_b128(&Ws[row * LDA + col], src);
        }
        stage_fence();
        __syncthreads();

        // hoist all fragment loads so ds_loads batch under one wait, then
        // issue the 4 WMMAs back-to-back
        const __bf16* arow = &As[(wave * 16 + ln) * LDA];
        v16bf af = ld8x2(arow + hi * 16, arow + hi * 16 + 8);
        v16bf bfrag[4];
#pragma unroll
        for (int t = 0; t < 4; ++t)
            bfrag[t] = ld16(&Ws[(t * 16 + ln) * LDA + hi * 16]);
#pragma unroll
        for (int t = 0; t < 4; ++t)
            acc[t] = wmma_bf16(af, bfrag[t], acc[t]);
    }

    // epilogue: lane holds col n = ln (+tile), rows m = hi*8 + r
#pragma unroll
    for (int t = 0; t < 4; ++t) {
        int o = o_blk + t * 16 + ln;
        float bv = bias[o];
        if (MODE == 0) {
            int s = o >> 10, rem = o & 1023;
            int h = rem >> 6, d = rem & 63;
            __bf16* qkv = (__bf16*)out;
#pragma unroll
            for (int r = 0; r < 8; ++r) {
                int gm = m_blk + wave * 16 + hi * 8 + r;
                if (gm < M) {
                    int b = gm / NTOK;
                    int n = gm - b * NTOK;
                    size_t idx = ((((size_t)s * BB + b) * HH + h) * NTOK + n) * HD + d;
                    qkv[idx] = (__bf16)(acc[t][r] + bv);
                }
            }
        } else {
            float* fo = (float*)out;
#pragma unroll
            for (int r = 0; r < 8; ++r) {
                int gm = m_blk + wave * 16 + hi * 8 + r;
                if (gm < M) fo[(size_t)gm * CC + o] = acc[t][r] + bv;
            }
        }
    }
}

// ---------------------------------------------------------------------------
// Kernel 2: LayerNorm(hd=64) + split RoPE on q and k, in place (bf16).
// One wave32 per (which, b, h, n) row; lane holds elems d and d+32.
// All reductions via DPP + permlanex16 (no LDS traffic).
// ---------------------------------------------------------------------------
__global__ __launch_bounds__(256)
void ln_rope_kernel(__bf16* __restrict__ qkv,
                    const float* __restrict__ patch_rope, // [1024][32]
                    const float* __restrict__ reg_rope,   // [4][32]
                    const float* __restrict__ qn_w, const float* __restrict__ qn_b,
                    const float* __restrict__ kn_w, const float* __restrict__ kn_b) {
    const long long rows_per = (long long)BB * HH * NTOK;
    long long row = (long long)blockIdx.x * 8 + (threadIdx.x >> 5);
    const int lane = threadIdx.x & 31;
    if (row >= 2 * rows_per) return;
    const int which = (row >= rows_per) ? 1 : 0;   // 0=q, 1=k
    long long r = row - (long long)which * rows_per;
    int n = (int)(r % NTOK);

    __bf16* p = qkv + row * HD;                    // q rows first, then k rows
    float v0 = (float)p[lane];
    float v1 = (float)p[lane + 32];

    float s = rowsum16(v0 + v1);
    s += xhalf(s);
    float mean = s * (1.0f / 64.0f);
    float d0 = v0 - mean, d1 = v1 - mean;
    float vs = rowsum16(d0 * d0 + d1 * d1);
    vs += xhalf(vs);
    float inv = rsqrtf(vs * (1.0f / 64.0f) + 1e-6f);

    const float* wsc = which ? kn_w : qn_w;
    const float* bsc = which ? kn_b : qn_b;
    v0 = d0 * inv * wsc[lane]      + bsc[lane];
    v1 = d1 * inv * wsc[lane + 32] + bsc[lane + 32];

    float o0 = v0, o1 = v1;
    if (n != 4) {                                  // token 4 (CLS) unrotated
        float ang = (n < 4) ? reg_rope[n * 32 + lane]
                            : patch_rope[(size_t)(n - 5) * 32 + lane];
        float c = __cosf(ang), si = __sinf(ang);
        o0 = v0 * c - v1 * si;
        o1 = v0 * si + v1 * c;
    }
    p[lane]      = (__bf16)o0;
    p[lane + 32] = (__bf16)o1;
}

// ---------------------------------------------------------------------------
// Kernel 3: flash attention per (b,h). 128 threads = 4 waves; wave owns 16
// query rows; key tiles of 32; online softmax (DPP row reductions); WMMA for
// QK^T and PV. Writes bf16 [B*N][1024] (head-interleaved) for the proj GEMM.
// ---------------------------------------------------------------------------
__global__ __launch_bounds__(128)
void attn_kernel(const __bf16* __restrict__ q, const __bf16* __restrict__ k,
                 const __bf16* __restrict__ v, __bf16* __restrict__ outb) {
    const int bh   = blockIdx.y;                   // 0..255
    const int b    = bh >> 4, h = bh & 15;
    const int wave = threadIdx.x >> 5, lane = threadIdx.x & 31;
    const int ln   = lane & 15, hi = lane >> 4;

    __shared__ __bf16 kt_s[32 * 64];               // [key][hd]
    __shared__ __bf16 vt_s[64 * 32];               // transposed [hd][key]
    __shared__ __bf16 p_s[4][16 * 32];             // per-wave P staging

    const size_t base = (size_t)bh * NTOK * HD;
    const int qrow0 = blockIdx.x * 64 + wave * 16;

    // Q fragments for K=0..31 and K=32..63 (A-matrix 16x32 layout)
    v16bf qf0 = {}, qf1 = {};
    {
        int qr = qrow0 + ln;
        if (qr < NTOK) {
            const __bf16* qp = q + base + (size_t)qr * HD;
            qf0 = ld8x2(qp + hi * 16,      qp + hi * 16 + 8);
            qf1 = ld8x2(qp + 32 + hi * 16, qp + 40 + hi * 16);
        }
    }

    float rowmax[8], rowsum[8];
    v8f o[4];
#pragma unroll
    for (int r = 0; r < 8; ++r) { rowmax[r] = -1e30f; rowsum[r] = 0.f; }
#pragma unroll
    for (int t = 0; t < 4; ++t) o[t] = (v8f){};
    const float scale = 0.125f;                    // 64^-0.5

    for (int kt0 = 0; kt0 < NTOK; kt0 += 32) {
        __syncthreads();
        { // cooperative load: K tile row-major (async), V tile transposed
            int row = threadIdx.x >> 2;            // 0..31
            int col = (threadIdx.x & 3) * 16;      // 0,16,32,48
            int gr  = kt0 + row;
            if (gr < NTOK) {
                const __bf16* kp = k + base + (size_t)gr * HD + col;
                const __bf16* vp = v + base + (size_t)gr * HD + col;
                stage_b128(&kt_s[row * 64 + col],     kp);
                stage_b128(&kt_s[row * 64 + col + 8], kp + 8);
                __bf16 tv[16];
                *(bf16x8*)tv       = *(const bf16x8*)vp;
                *(bf16x8*)(tv + 8) = *(const bf16x8*)(vp + 8);
#pragma unroll
                for (int i = 0; i < 16; ++i) vt_s[(col + i) * 32 + row] = tv[i];
            } else {
#pragma unroll
                for (int i = 0; i < 16; ++i) {
                    kt_s[row * 64 + col + i]   = (__bf16)0.f;
                    vt_s[(col + i) * 32 + row] = (__bf16)0.f;
                }
            }
        }
        stage_fence();
        __syncthreads();

        // S = q @ k^T: hoist all 4 B-fragments, then 4 back-to-back WMMAs
        v16bf kb[4];
        kb[0] = ld16(&kt_s[ln * 64 + hi * 16]);
        kb[1] = ld16(&kt_s[ln * 64 + 32 + hi * 16]);
        kb[2] = ld16(&kt_s[(16 + ln) * 64 + hi * 16]);
        kb[3] = ld16(&kt_s[(16 + ln) * 64 + 32 + hi * 16]);
        v8f s0 = (v8f){}, s1 = (v8f){};
        s0 = wmma_bf16(qf0, kb[0], s0);
        s0 = wmma_bf16(qf1, kb[1], s0);
        s1 = wmma_bf16(qf0, kb[2], s1);
        s1 = wmma_bf16(qf1, kb[3], s1);

        const int c0 = kt0 + ln, c1 = kt0 + 16 + ln;
#pragma unroll
        for (int r = 0; r < 8; ++r) {
            float a = s0[r] * scale, bb = s1[r] * scale;
            if (c0 >= NTOK) a  = -1e30f;
            if (c1 >= NTOK) bb = -1e30f;
            // row max over the 16 lanes of this half (pure-VALU DPP tree)
            float mx    = rowmax16(fmaxf(a, bb));
            float mnew  = fmaxf(rowmax[r], mx);
            float alpha = __expf(rowmax[r] - mnew);
            float p0 = __expf(a - mnew), p1 = __expf(bb - mnew);
            float ps = rowsum16(p0 + p1);
            rowsum[r] = rowsum[r] * alpha + ps;
            rowmax[r] = mnew;
#pragma unroll
            for (int t = 0; t < 4; ++t) o[t][r] = o[t][r] * alpha;
            int mrow = hi * 8 + r;
            p_s[wave][mrow * 32 + ln]      = (__bf16)p0;
            p_s[wave][mrow * 32 + 16 + ln] = (__bf16)p1;
        }

        // P (16x32) as A-fragment, PV: hoist V fragments then 4 WMMAs
        const __bf16* pr = &p_s[wave][ln * 32];
        v16bf pf = ld8x2(pr + hi * 16, pr + hi * 16 + 8);
        v16bf vb[4];
#pragma unroll
        for (int t = 0; t < 4; ++t)
            vb[t] = ld16(&vt_s[(t * 16 + ln) * 32 + hi * 16]);
#pragma unroll
        for (int t = 0; t < 4; ++t)
            o[t] = wmma_bf16(pf, vb[t], o[t]);
    }

    // normalize + store bf16 [B*N][C] with head interleave
#pragma unroll
    for (int t = 0; t < 4; ++t) {
        int d = h * HD + t * 16 + ln;
#pragma unroll
        for (int r = 0; r < 8; ++r) {
            int qr = qrow0 + hi * 8 + r;
            if (qr < NTOK)
                outb[((size_t)b * NTOK + qr) * CC + d] =
                    (__bf16)(o[t][r] / rowsum[r]);
        }
    }
}

// ---------------------------------------------------------------------------
// Host launcher
// ---------------------------------------------------------------------------
static inline size_t align256(size_t x) { return (x + 255) & ~(size_t)255; }

extern "C" void kernel_launch(void* const* d_in, const int* in_sizes, int n_in,
                              void* d_out, int out_size, void* d_ws, size_t ws_size,
                              hipStream_t stream) {
    (void)in_sizes; (void)n_in; (void)out_size; (void)ws_size;
    const float* x          = (const float*)d_in[0];
    const float* patch_rope = (const float*)d_in[1];
    const float* reg_rope   = (const float*)d_in[2];
    const float* qkv_w      = (const float*)d_in[3];
    const float* qkv_b      = (const float*)d_in[4];
    const float* proj_w     = (const float*)d_in[5];
    const float* proj_b     = (const float*)d_in[6];
    const float* qn_w       = (const float*)d_in[7];
    const float* qn_b       = (const float*)d_in[8];
    const float* kn_w       = (const float*)d_in[9];
    const float* kn_b       = (const float*)d_in[10];

    // workspace layout (bf16): x/attn (reused), qkv_w, proj_w, qkv[3][B][H][N][64]
    char* ws = (char*)d_ws;
    size_t sz_x  = (size_t)MTOT * CC * sizeof(__bf16);
    size_t sz_qw = (size_t)3 * CC * CC * sizeof(__bf16);
    size_t sz_pw = (size_t)CC * CC * sizeof(__bf16);
    size_t o_x   = 0;
    size_t o_qw  = align256(o_x + sz_x);
    size_t o_pw  = align256(o_qw + sz_qw);
    size_t o_qkv = align256(o_pw + sz_pw);
    __bf16* x_bf   = (__bf16*)(ws + o_x);    // reused as attention output
    __bf16* qw_bf  = (__bf16*)(ws + o_qw);
    __bf16* pw_bf  = (__bf16*)(ws + o_pw);
    __bf16* qkv_bf = (__bf16*)(ws + o_qkv);
    const size_t per = (size_t)BB * HH * NTOK * HD;
    __bf16* q_bf = qkv_bf;
    __bf16* k_bf = qkv_bf + per;
    __bf16* v_bf = qkv_bf + 2 * per;

    // 0) conversions to bf16
    cvt_bf16_kernel<<<4096, 256, 0, stream>>>(x, x_bf, (long long)MTOT * CC);
    cvt_bf16_kernel<<<2048, 256, 0, stream>>>(qkv_w, qw_bf, (long long)3 * CC * CC);
    cvt_bf16_kernel<<<1024, 256, 0, stream>>>(proj_w, pw_bf, (long long)CC * CC);

    // 1) QKV GEMM -> scattered [3][B][H][N][64] bf16
    dim3 g1((MTOT + 127) / 128, (3 * CC) / 64);
    gemm_bf16_kernel<0><<<g1, 256, 0, stream>>>(x_bf, qw_bf, qkv_b,
                                                (void*)qkv_bf, MTOT);

    // 2) LayerNorm + RoPE on q,k (in place)
    long long rows = 2LL * BB * HH * NTOK;
    ln_rope_kernel<<<(unsigned)((rows + 7) / 8), 256, 0, stream>>>(
        qkv_bf, patch_rope, reg_rope, qn_w, qn_b, kn_w, kn_b);

    // 3) flash attention -> bf16 [B*N][C] (reuses x_bf buffer)
    dim3 g3((NTOK + 63) / 64, BB * HH);
    attn_kernel<<<g3, 128, 0, stream>>>(q_bf, k_bf, v_bf, x_bf);

    // 4) output projection -> f32 d_out [B][N][C]
    dim3 g4((MTOT + 127) / 128, CC / 64);
    gemm_bf16_kernel<1><<<g4, 256, 0, stream>>>(x_bf, pw_bf, proj_b,
                                                d_out, MTOT);
}
